// DeltaNet_3582002724932
// MI455X (gfx1250) — matrix operand order repaired and checked by reading, not compiled, not run on hardware
//
#include <hip/hip_runtime.h>
#include <cstdint>
#include <cstddef>

#define B_ 4
#define L_ 2048
#define D_ 1024
#define H_ 4
#define DK_ 128
#define DV_ 256

typedef __attribute__((ext_vector_type(16))) __bf16 v16bf;
typedef __attribute__((ext_vector_type(2)))  __bf16 v2bf;
typedef __attribute__((ext_vector_type(2)))  float  v2f;
typedef __attribute__((ext_vector_type(8)))  float  v8f;

// ---------- optional gfx1250 async global->LDS path ----------
#if __has_builtin(__builtin_amdgcn_global_load_async_to_lds_b128)
#define ASYNC_LDS 1
#else
#define ASYNC_LDS 0
#endif

#if ASYNC_LDS
typedef int v4i_vs __attribute__((vector_size(16)));
typedef __attribute__((address_space(1))) v4i_vs* gv4i_p;   // global AS pointer to v4i
typedef __attribute__((address_space(3))) v4i_vs* lv4i_p;   // LDS AS pointer to v4i
__device__ __forceinline__ void async_copy16(const void* g, void* l) {
  __builtin_amdgcn_global_load_async_to_lds_b128((gv4i_p)(void*)g, (lv4i_p)l, 0, 0);
}
__device__ __forceinline__ void async_wait0() {
#if __has_builtin(__builtin_amdgcn_s_wait_asynccnt)
  __builtin_amdgcn_s_wait_asynccnt(0);
#else
  asm volatile("s_wait_asynccnt 0" ::: "memory");
#endif
}
#endif

// ---------- helpers ----------
__device__ __forceinline__ __bf16 to_bf(float x) { return (__bf16)x; }

__device__ __forceinline__ unsigned pack2(float lo, float hi) {
  v2f f; f[0] = lo; f[1] = hi;
  union { v2bf b; unsigned u; } p;
  p.b = __builtin_convertvector(f, v2bf);     // packed f32x2 -> bf16x2
  return p.u;
}

__device__ __forceinline__ v8f wmma_bf16(const v16bf& a, const v16bf& b, const v8f& c) {
  return __builtin_amdgcn_wmma_f32_16x16x32_bf16(false, a, false, b, (short)0, c, false, false);
}

// A fragment from row-major packed-pair array base[m*ldp + pair] (ldp % 4 == 0
// -> two 16B-aligned ds_load_b128 per fragment).
__device__ __forceinline__ v16bf frag_a_pk(const unsigned* base, int ldp, int row0, int kp0) {
  const int lane = threadIdx.x & 31;
  const int m = row0 + (lane & 15);
  const int hh = lane >> 4;
  union { v16bf bf; unsigned u[8]; } r;
#pragma unroll
  for (int v = 0; v < 8; ++v)
    r.u[v] = base[m * ldp + kp0 + hh * 8 + v];
  return r.bf;
}

// B fragment, operand X (KxN) stored transposed-packed: base[n*ldp + kpair].
__device__ __forceinline__ v16bf frag_bT_pk(const unsigned* base, int ldp, int kp0, int col0) {
  const int lane = threadIdx.x & 31;
  const int n = col0 + (lane & 15);
  const int hh = lane >> 4;
  union { v16bf bf; unsigned u[8]; } r;
#pragma unroll
  for (int v = 0; v < 8; ++v)
    r.u[v] = base[n * ldp + kp0 + hh * 8 + v];
  return r.bf;
}

__device__ __forceinline__ v8f load_c(const float* base, int ld, int row0, int col0) {
  const int lane = threadIdx.x & 31;
  const int n  = col0 + (lane & 15);
  const int m0 = row0 + (lane >> 4) * 8;
  v8f c;
#pragma unroll
  for (int r = 0; r < 8; ++r) c[r] = base[(m0 + r) * ld + n];
  return c;
}

// ---------- kernel 1: causal depthwise conv (K=4) + SiLU -> packed bf16 h ----------
__global__ __launch_bounds__(256) void conv_silu_kernel(const float* __restrict__ x,
                                                        const float* __restrict__ w,
                                                        unsigned* __restrict__ hp) {
  const size_t idx = (size_t)blockIdx.x * 256 + threadIdx.x;  // over B*L*D/2
  const int dp = (int)(idx & (D_ / 2 - 1));
  const int d = 2 * dp;
  const size_t bl = idx >> 9;
  const int l = (int)(bl & (L_ - 1));
  const size_t brow = bl - (size_t)l;
  float a0 = 0.f, a1 = 0.f;
#pragma unroll
  for (int j = 0; j < 4; ++j) {
    const int ls = l - 3 + j;
    if (ls >= 0) {
      const float* xr = &x[(brow + (size_t)ls) * D_ + d];
      a0 += xr[0] * w[d * 4 + j];
      a1 += xr[1] * w[(d + 1) * 4 + j];
    }
  }
  hp[idx] = pack2(a0 / (1.f + __expf(-a0)), a1 / (1.f + __expf(-a1)));
}

// ---------- kernel 1b: one-time weight prep: Wt[n][K/2] = packed-bf16 W^T pairs ----------
__global__ __launch_bounds__(256) void wprep_kernel(const float* __restrict__ W,
                                                    unsigned* __restrict__ Wt, int N) {
  const int idx = blockIdx.x * 256 + threadIdx.x;   // over N * 512 ; n fastest (coalesced reads)
  const int n = idx % N;
  const int p = idx / N;
  Wt[(size_t)n * 512 + p] = pack2(W[(size_t)(2 * p) * N + n], W[(size_t)(2 * p + 1) * N + n]);
}

// ---------- kernel 2: GEMM, packed-bf16 in / f32 out ----------
// C[MxN] = A[MxK] @ B[KxN]; A packed pairs [m][K/2]; B transposed-packed [n][K/2].
// Tile 128x128, 8 waves each 32x64. Staging: async global->LDS b128 if available.
#define GTM 128
#define GTN 128
#define GLDP 20
__global__ __launch_bounds__(256) void gemm_pk_wmma(const unsigned* __restrict__ Apk,
                                                    const unsigned* __restrict__ Btg,
                                                    float* __restrict__ C,
                                                    int M, int N, int Kp) {   // Kp = K/2 pairs
  __shared__ unsigned Asp[GTM][GLDP];
  __shared__ unsigned Btp[GTN][GLDP];
  const int tid  = threadIdx.x;
  const int wave = tid >> 5;
  const int tm = blockIdx.x * GTM;
  const int tn = blockIdx.y * GTN;
  const int row0 = (wave >> 1) * 32;
  const int colb = (wave & 1) * 64;
  const int qd = (tid & 3) * 4;
  const int r0 = tid >> 2;
  v8f acc[2][4] = {};
  for (int kp0 = 0; kp0 < Kp; kp0 += 16) {
#pragma unroll
    for (int s = 0; s < 2; ++s) {
      const int r = r0 + 64 * s;
#if ASYNC_LDS
      async_copy16(&Apk[(size_t)(tm + r) * Kp + kp0 + qd], &Asp[r][qd]);
      async_copy16(&Btg[(size_t)(tn + r) * Kp + kp0 + qd], &Btp[r][qd]);
#else
      *(uint4*)&Asp[r][qd] = *(const uint4*)&Apk[(size_t)(tm + r) * Kp + kp0 + qd];
      *(uint4*)&Btp[r][qd] = *(const uint4*)&Btg[(size_t)(tn + r) * Kp + kp0 + qd];
#endif
    }
    if (kp0 + 16 < Kp) {   // gfx1250 global_prefetch of next tiles
      __builtin_prefetch(&Apk[(size_t)(tm + (tid >> 1)) * Kp + kp0 + 16 + (tid & 1) * 8], 0, 1);
      __builtin_prefetch(&Btg[(size_t)(tn + (tid >> 1)) * Kp + kp0 + 16 + (tid & 1) * 8], 0, 1);
    }
#if ASYNC_LDS
    async_wait0();
#endif
    __syncthreads();
    v16bf a0 = frag_a_pk(&Asp[0][0], GLDP, row0, 0);
    v16bf a1 = frag_a_pk(&Asp[0][0], GLDP, row0 + 16, 0);
#pragma unroll
    for (int t = 0; t < 4; ++t) {
      v16bf bf = frag_bT_pk(&Btp[0][0], GLDP, 0, colb + 16 * t);
      acc[0][t] = wmma_bf16(a0, bf, acc[0][t]);
      acc[1][t] = wmma_bf16(a1, bf, acc[1][t]);
    }
    __syncthreads();
  }
  const int lane = tid & 31;
  const int n_in = lane & 15;
#pragma unroll
  for (int rr = 0; rr < 2; ++rr) {
    const int m0 = row0 + rr * 16 + (lane >> 4) * 8;
#pragma unroll
    for (int t = 0; t < 4; ++t)
#pragma unroll
      for (int r = 0; r < 8; ++r)
        C[(size_t)(tm + m0 + r) * N + tn + colb + 16 * t + n_in] = acc[rr][t][r];
  }
}

// ---------- kernel 3: beta = sigmoid(h @ Wbeta), N=4 (reads bf16 h) ----------
__global__ __launch_bounds__(128) void beta_kernel(const unsigned* __restrict__ hp,
                                                   const float* __restrict__ Wb,
                                                   float* __restrict__ beta) {
  __shared__ float red[4][128];
  const __bf16* hb = (const __bf16*)hp;
  const int m = blockIdx.x;
  const int t = threadIdx.x;
  float s[4] = {0.f, 0.f, 0.f, 0.f};
  for (int k = t; k < D_; k += 128) {
    const float x = (float)hb[(size_t)m * D_ + k];
#pragma unroll
    for (int hh = 0; hh < 4; ++hh) s[hh] += x * Wb[k * 4 + hh];
  }
#pragma unroll
  for (int hh = 0; hh < 4; ++hh) red[hh][t] = s[hh];
  __syncthreads();
  for (int o = 64; o > 0; o >>= 1) {
    if (t < o) {
#pragma unroll
      for (int hh = 0; hh < 4; ++hh) red[hh][t] += red[hh][t + o];
    }
    __syncthreads();
  }
  if (t < 4) beta[(size_t)m * 4 + t] = 1.f / (1.f + __expf(-red[t][0]));
}

// ---------- kernel 4: L2-normalize k rows, scale q by DK^-0.5 ----------
__global__ __launch_bounds__(128) void qk_norm_kernel(float* __restrict__ q, float* __restrict__ k) {
  __shared__ float red[128];
  const int row = blockIdx.x >> 2;
  const int hh  = blockIdx.x & 3;
  const int d   = threadIdx.x;
  const size_t off = (size_t)row * 512 + hh * 128 + d;
  const float kv = k[off];
  red[d] = kv * kv;
  __syncthreads();
  for (int o = 64; o > 0; o >>= 1) {
    if (d < o) red[d] += red[d + o];
    __syncthreads();
  }
  k[off] = kv * rsqrtf(red[0] + 1e-12f);
  q[off] *= 0.08838834764831844f;
}

// ---------- kernel 5: chunked delta rule (C=32) ----------
#define CCH 32
#define OFF_S    0                              // f32 [128][264]
#define OFF_SP   (OFF_S   + 128 * 264 * 4)      // uint [256][68] S^T bf16 shadow
#define OFF_KCP  (OFF_SP  + 256 * 68 * 4)       // uint [32][68]
#define OFF_QCP  (OFF_KCP + 32 * 68 * 4)        // uint [32][68]
#define OFF_KTP  (OFF_QCP + 32 * 68 * 4)        // uint [128][20]
#define OFF_VCP  (OFF_KTP + 128 * 20 * 4)       // uint [256][20]
#define OFF_UCP  (OFF_VCP + 256 * 20 * 4)       // uint [256][20]
#define OFF_WCP  (OFF_UCP + 256 * 20 * 4)       // uint [32][68] (bf16 view ld 136)
#define OFF_TBP  (OFF_WCP + 32 * 68 * 4)        // uint [32][20] (bf16 view ld 40)
#define OFF_MMP  (OFF_TBP + 32 * 20 * 4)        // uint [32][20]
#define OFF_AM   (OFF_MMP + 32 * 20 * 4)        // f32 [32][36]
#define OFF_TM   (OFF_AM  + 32 * 36 * 4)        // f32 [32][36]
#define OFF_BC   (OFF_TM  + 32 * 36 * 4)        // f32 [32]
#define SCAN_LDS_BYTES (OFF_BC + 32 * 4)        // = 296576

__global__ __launch_bounds__(256) void delta_scan_kernel(const float* __restrict__ qg,
                                                         const float* __restrict__ kg,
                                                         const float* __restrict__ vg,
                                                         const float* __restrict__ bg,
                                                         float* __restrict__ og) {
  extern __shared__ unsigned char smraw[];
  float*    S   = (float*)(smraw + OFF_S);
  unsigned* Sp  = (unsigned*)(smraw + OFF_SP);
  unsigned* Kcp = (unsigned*)(smraw + OFF_KCP);
  unsigned* Qcp = (unsigned*)(smraw + OFF_QCP);
  unsigned* KTp = (unsigned*)(smraw + OFF_KTP);
  unsigned* Vcp = (unsigned*)(smraw + OFF_VCP);
  unsigned* Ucp = (unsigned*)(smraw + OFF_UCP);
  unsigned* Wcp = (unsigned*)(smraw + OFF_WCP);
  __bf16*   Wcb = (__bf16*)Wcp;
  unsigned* Tbp = (unsigned*)(smraw + OFF_TBP);
  __bf16*   Tbb = (__bf16*)Tbp;
  unsigned* Mmp = (unsigned*)(smraw + OFF_MMP);
  __bf16*   Mmb = (__bf16*)Mmp;
  float*    Am  = (float*)(smraw + OFF_AM);
  float*    Tm  = (float*)(smraw + OFF_TM);
  float*    bc  = (float*)(smraw + OFF_BC);

  const int bh = blockIdx.x;
  const int b  = bh >> 2;
  const int hh = bh & 3;
  const int tid  = threadIdx.x;
  const int wave = tid >> 5;
  const int lane = tid & 31;

  const float* qp = qg + (size_t)b * L_ * 512  + hh * DK_;
  const float* kp = kg + (size_t)b * L_ * 512  + hh * DK_;
  const float* vp = vg + (size_t)b * L_ * 1024 + hh * DV_;
  const float* bp = bg + (size_t)b * L_ * 4    + hh;
  float*       op = og + (size_t)b * L_ * 1024 + hh * DV_;

  for (int i = tid; i < 128 * 264; i += 256) S[i] = 0.f;
  for (int i = tid; i < 256 * 68; i += 256) Sp[i] = 0u;
  __syncthreads();

  const int sc_p  = tid & 63,  sc_c  = tid >> 6;
  const int sk_d  = tid & 127, sk_p  = tid >> 7;
  const int sv_n  = tid;

  for (int c0 = 0; c0 < L_; c0 += CCH) {
    // ---- stage chunk into packed-bf16 LDS (strength-reduced) ----
    {
      const float* kr = kp + (size_t)(c0 + sc_c) * 512 + 2 * sc_p;
      const float* qr = qp + (size_t)(c0 + sc_c) * 512 + 2 * sc_p;
      unsigned* kd = Kcp + sc_c * 68 + sc_p;
      unsigned* qd = Qcp + sc_c * 68 + sc_p;
#pragma unroll
      for (int s = 0; s < 8; ++s) {
        kd[s * 4 * 68] = pack2(kr[s * 2048], kr[s * 2048 + 1]);
        qd[s * 4 * 68] = pack2(qr[s * 2048], qr[s * 2048 + 1]);
      }
    }
    {
      const float* kc = kp + (size_t)(c0 + 2 * sk_p) * 512 + sk_d;
      unsigned* kt = KTp + sk_d * 20 + sk_p;
#pragma unroll
      for (int s = 0; s < 8; ++s)
        kt[2 * s] = pack2(kc[s * 2048], kc[s * 2048 + 512]);
    }
    {
      const float* vc = vp + (size_t)c0 * 1024 + sv_n;
      unsigned* vd = Vcp + sv_n * 20;
#pragma unroll
      for (int s = 0; s < 16; ++s)
        vd[s] = pack2(vc[(size_t)(2 * s) * 1024], vc[(size_t)(2 * s + 1) * 1024]);
    }
    if (tid < CCH) bc[tid] = bp[(size_t)(c0 + tid) * 4];
    __syncthreads();

    // ---- waves 0-3: Am = diag(b)*stril(K K^T); waves 4-7: Mmp = stril(Q K^T) ----
    {
      const int w4 = wave & 3;
      const int row0 = (w4 & 1) * 16;
      const int col0 = (w4 >> 1) * 16;
      v8f acc = {};
      const unsigned* Ap = (wave < 4) ? Kcp : Qcp;
      for (int kk2 = 0; kk2 < 64; kk2 += 16) {
        v16bf a  = frag_a_pk(Ap, 68, row0, kk2);
        v16bf bb = frag_bT_pk(Kcp, 68, kk2, col0);
        acc = wmma_bf16(a, bb, acc);
      }
      const int n  = col0 + (lane & 15);
      const int m0 = row0 + (lane >> 4) * 8;
      if (wave < 4) {
#pragma unroll
        for (int r = 0; r < 8; ++r)
          Am[(m0 + r) * 36 + n] = ((m0 + r) > n) ? acc[r] * bc[m0 + r] : 0.f;
      } else {
#pragma unroll
        for (int r = 0; r < 8; ++r)
          Mmb[(m0 + r) * 40 + n] = to_bf(((m0 + r) > n) ? acc[r] : 0.f);
      }
    }
    __syncthreads();

    // ---- T = (I+Am)^-1 (wave 0); T' = T*diag(b) in bf16 ----
    if (wave == 0) {
      const int cc = lane;
      const float bcc = bc[cc];
      for (int i = 0; i < CCH; ++i) {
        float t = (cc == i) ? 1.f : 0.f;
        for (int j = 0; j < i; ++j) t -= Am[i * 36 + j] * Tm[j * 36 + cc];
        Tm[i * 36 + cc] = t;
        Tbb[i * 40 + cc] = to_bf(t * bcc);
      }
    }
    __syncthreads();

    // ---- W = -(T' @ K)  (32x128, K=32) ----
    {
      const int row0 = (wave >> 2) * 16;
      const int cb   = (wave & 3) * 32;
      v16bf a = frag_a_pk(Tbp, 20, row0, 0);
#pragma unroll
      for (int t = 0; t < 2; ++t) {
        const int col0 = cb + t * 16;
        v16bf bb = frag_bT_pk(KTp, 20, 0, col0);
        v8f acc = {};
        acc = wmma_bf16(a, bb, acc);
        const int n  = col0 + (lane & 15);
        const int m0 = row0 + (lane >> 4) * 8;
#pragma unroll
        for (int r = 0; r < 8; ++r) Wcb[(m0 + r) * 136 + n] = to_bf(-acc[r]);
      }
    }
    __syncthreads();

    // ---- U = T'@V + W@S  (32x256) -> U^T packed ----
    {
      const int row0 = (wave >> 2) * 16;
      const int cb   = (wave & 3) * 64;
      v16bf at = frag_a_pk(Tbp, 20, row0, 0);
#pragma unroll
      for (int t = 0; t < 4; ++t) {
        const int col0 = cb + t * 16;
        v8f acc = {};
        acc = wmma_bf16(at, frag_bT_pk(Vcp, 20, 0, col0), acc);
        for (int kk2 = 0; kk2 < 64; kk2 += 16) {
          v16bf aw = frag_a_pk(Wcp, 68, row0, kk2);
          v16bf bs = frag_bT_pk(Sp, 68, kk2, col0);
          acc = wmma_bf16(aw, bs, acc);
        }
        const int n  = col0 + (lane & 15);
        const int m0 = row0 + (lane >> 4) * 8;
        *(uint4*)&Ucp[n * 20 + (m0 >> 1)] =
            make_uint4(pack2(acc[0], acc[1]), pack2(acc[2], acc[3]),
                       pack2(acc[4], acc[5]), pack2(acc[6], acc[7]));
      }
    }
    __syncthreads();

    // ---- O = Q@S + Mm@U  -> global f32 ----
    {
      const int row0 = (wave >> 2) * 16;
      const int cb   = (wave & 3) * 64;
      v16bf am = frag_a_pk(Mmp, 20, row0, 0);
#pragma unroll
      for (int t = 0; t < 4; ++t) {
        const int col0 = cb + t * 16;
        v8f acc = {};
        for (int kk2 = 0; kk2 < 64; kk2 += 16) {
          v16bf aq = frag_a_pk(Qcp, 68, row0, kk2);
          v16bf bs = frag_bT_pk(Sp, 68, kk2, col0);
          acc = wmma_bf16(aq, bs, acc);
        }
        acc = wmma_bf16(am, frag_bT_pk(Ucp, 20, 0, col0), acc);
        const int n  = col0 + (lane & 15);
        const int m0 = row0 + (lane >> 4) * 8;
#pragma unroll
        for (int r = 0; r < 8; ++r)
          op[(size_t)(c0 + m0 + r) * 1024 + n] = acc[r];
      }
    }
    __syncthreads();

    // ---- S += K^T @ U  (128x256, K=32); f32 accumulate + packed S^T shadow ----
    {
      const int row0 = wave * 16;
      v16bf aK = frag_a_pk(KTp, 20, row0, 0);
#pragma unroll
      for (int t = 0; t < 16; ++t) {
        const int col0 = t * 16;
        v8f acc = load_c(S, 264, row0, col0);
        acc = wmma_bf16(aK, frag_bT_pk(Ucp, 20, 0, col0), acc);
        const int n  = col0 + (lane & 15);
        const int m0 = row0 + (lane >> 4) * 8;
#pragma unroll
        for (int r = 0; r < 8; ++r) S[(m0 + r) * 264 + n] = acc[r];
        *(uint4*)&Sp[n * 68 + (m0 >> 1)] =
            make_uint4(pack2(acc[0], acc[1]), pack2(acc[2], acc[3]),
                       pack2(acc[4], acc[5]), pack2(acc[6], acc[7]));
      }
    }
    __syncthreads();
  }
}

// ---------- kernel 6: RMS norm over DV + SiLU gate -> bf16 out ----------
__global__ __launch_bounds__(256) void rms_gate_kernel(const float* __restrict__ o,
                                                       const float* __restrict__ g,
                                                       const float* __restrict__ nw,
                                                       unsigned* __restrict__ outp) {
  __shared__ float red[256];
  __bf16* ob = (__bf16*)outp;
  const int row = blockIdx.x >> 2;
  const int hh  = blockIdx.x & 3;
  const int j   = threadIdx.x;
  const size_t off = (size_t)row * 1024 + hh * 256 + j;
  const float ov = o[off];
  red[j] = ov * ov;
  __syncthreads();
  for (int s = 128; s > 0; s >>= 1) {
    if (j < s) red[j] += red[j + s];
    __syncthreads();
  }
  const float scale = rsqrtf(red[0] * (1.f / 256.f) + 1e-5f);
  const float gv = g[off];
  ob[off] = to_bf(ov * scale * nw[j] * gv / (1.f + __expf(-gv)));
}

// ---------- host ----------
extern "C" void kernel_launch(void* const* d_in, const int* in_sizes, int n_in,
                              void* d_out, int out_size, void* d_ws, size_t ws_size,
                              hipStream_t stream) {
  (void)in_sizes; (void)n_in; (void)out_size; (void)ws_size;
  const float* x  = (const float*)d_in[0];
  const float* cw = (const float*)d_in[1];
  const float* Wq = (const float*)d_in[2];
  const float* Wk = (const float*)d_in[3];
  const float* Wv = (const float*)d_in[4];
  const float* Wb = (const float*)d_in[5];
  const float* Wg = (const float*)d_in[6];
  const float* Wo = (const float*)d_in[7];
  const float* nw = (const float*)d_in[8];
  float* out = (float*)d_out;

  const size_t M = (size_t)B_ * L_;   // 8192
  unsigned* h_pk  = (unsigned*)d_ws;                 // M*512 packed bf16
  float*    q_    = (float*)(h_pk + M * 512);
  float*    k_    = q_  + M * 512;
  float*    v_    = k_  + M * 512;
  float*    g_    = v_  + M * 1024;
  float*    be_   = g_  + M * 1024;
  float*    o_    = be_ + M * 4;
  unsigned* of_pk = (unsigned*)(o_ + M * 1024);      // M*512 packed bf16
  unsigned* wq_t  = of_pk + M * 512;                 // [512][512]
  unsigned* wk_t  = wq_t + 512 * 512;
  unsigned* wv_t  = wk_t + 512 * 512;                // [1024][512]
  unsigned* wg_t  = wv_t + 1024 * 512;
  unsigned* wo_t  = wg_t + 1024 * 512;

  (void)hipFuncSetAttribute((const void*)delta_scan_kernel,
                            hipFuncAttributeMaxDynamicSharedMemorySize, SCAN_LDS_BYTES);

  conv_silu_kernel<<<(int)(M * D_ / 2 / 256), 256, 0, stream>>>(x, cw, h_pk);
  wprep_kernel<<<512 * 512 / 256, 256, 0, stream>>>(Wq, wq_t, 512);
  wprep_kernel<<<512 * 512 / 256, 256, 0, stream>>>(Wk, wk_t, 512);
  wprep_kernel<<<1024 * 512 / 256, 256, 0, stream>>>(Wv, wv_t, 1024);
  wprep_kernel<<<1024 * 512 / 256, 256, 0, stream>>>(Wg, wg_t, 1024);
  wprep_kernel<<<1024 * 512 / 256, 256, 0, stream>>>(Wo, wo_t, 1024);

  dim3 g512((unsigned)(M / GTM), 4);
  dim3 g1024((unsigned)(M / GTM), 8);
  gemm_pk_wmma<<<g512,  256, 0, stream>>>(h_pk, wq_t, q_, (int)M, 512, 512);
  gemm_pk_wmma<<<g512,  256, 0, stream>>>(h_pk, wk_t, k_, (int)M, 512, 512);
  gemm_pk_wmma<<<g1024, 256, 0, stream>>>(h_pk, wv_t, v_, (int)M, 1024, 512);
  gemm_pk_wmma<<<g1024, 256, 0, stream>>>(h_pk, wg_t, g_, (int)M, 1024, 512);

  beta_kernel<<<(int)M, 128, 0, stream>>>(h_pk, Wb, be_);
  qk_norm_kernel<<<(int)(M * H_), 128, 0, stream>>>(q_, k_);

  delta_scan_kernel<<<B_ * H_, 256, SCAN_LDS_BYTES, stream>>>(q_, k_, v_, be_, o_);

  rms_gate_kernel<<<(int)(M * H_), 256, 0, stream>>>(o_, g_, nw, of_pk);
  gemm_pk_wmma<<<g1024, 256, 0, stream>>>(of_pk, wo_t, out, (int)M, 1024, 512);
}
